// SELM_9887014715908
// MI455X (gfx1250) — matrix operand order repaired and checked by reading, not compiled, not run on hardware
//
#include <hip/hip_runtime.h>
#include <hip/hip_bf16.h>

typedef __bf16 bf16_t;
typedef __attribute__((ext_vector_type(16))) __bf16 v16bf;
typedef __attribute__((ext_vector_type(8)))  float  v8f;
typedef __attribute__((ext_vector_type(2)))  float  v2f;

static __device__ __forceinline__ unsigned int pack_bf16x2(float lo, float hi) {
  union { bf16_t h[2]; unsigned int u; } u;
  u.h[0] = (bf16_t)lo;
  u.h[1] = (bf16_t)hi;
  return u.u;
}

// fragment = 8 packed uint32 (16 bf16) loaded as two aligned int4
union FragU { int4 i[2]; v16bf v; };

struct TileRegs { float4 a0, a1, u0, u1, w0, w1; };

// ---------------------------------------------------------------------------
// GEMM: C = relu(A[M,K] @ W[K,N] + bias[N]) (f32 memory, bf16 WMMA compute).
// Block 256 threads (8 waves), tile 64(M) x 128(N), K-step 32.
// LDS tiles stored PRE-PACKED IN FRAGMENT ORDER (each lane's WMMA fragment is
// two aligned ds_load_b128). Double-buffered LDS + register prefetch:
// one barrier per K-step, global loads overlap WMMA compute.
// gridDim.z > 1 => split-K: raw f32 atomicAdd into zeroed C, epilogue later.
// ---------------------------------------------------------------------------
__global__ __launch_bounds__(256) void gemm_bias_relu(
    const float* __restrict__ A, const float* __restrict__ W,
    const float* __restrict__ bias, float* __restrict__ C,
    int M, int N, int K)
{
  // packed bf16-pair tiles, pitch 20 dwords (80B: 16B-aligned, bank-clean)
  __shared__ __align__(16) unsigned int As_p[2][64][20];    // [buf][row][kp]
  __shared__ __align__(16) unsigned int Bs_p[2][128][20];   // [buf][col][kp]

  const int t    = threadIdx.x;
  const int lane = t & 31;
  const int wave = t >> 5;
  const int wm   = wave >> 1;          // 0..3 -> 16-row sub-tile
  const int wn   = wave & 1;           // 0..1 -> 64-col half
  const int m_l  = lane & 15;
  const int kh   = lane >> 4;

  const int rowBase = blockIdx.y * 64;
  const int colBase = blockIdx.x * 128;

  const int kChunk = K / gridDim.z;
  const int kBeg   = blockIdx.z * kChunk;
  const int nIter  = kChunk / 32;

  // staging coords (constant across k-steps)
  const int a_r   = t >> 2;            // 0..63
  const int a_kp0 = (t & 3) * 4;       // 0,4,8,12  (pair index)
  const int b_kp  = t >> 4;            // 0..15
  const int b_c0  = (t & 15) * 8;      // 0..120

  v8f acc0 = {}, acc1 = {}, acc2 = {}, acc3 = {};

  auto load_tile = [&](int k0, TileRegs& r) {
    const float4* s4 = (const float4*)(A + (long long)(rowBase + a_r) * K + k0 + a_kp0 * 2);
    r.a0 = s4[0]; r.a1 = s4[1];
    const float4* p0 = (const float4*)(W + (long long)(k0 + 2 * b_kp)     * N + colBase + b_c0);
    const float4* p1 = (const float4*)(W + (long long)(k0 + 2 * b_kp + 1) * N + colBase + b_c0);
    r.u0 = p0[0]; r.u1 = p0[1];
    r.w0 = p1[0]; r.w1 = p1[1];
  };
  auto stage_tile = [&](int buf, const TileRegs& r) {
    int4 p;
    p.x = (int)pack_bf16x2(r.a0.x, r.a0.y);
    p.y = (int)pack_bf16x2(r.a0.z, r.a0.w);
    p.z = (int)pack_bf16x2(r.a1.x, r.a1.y);
    p.w = (int)pack_bf16x2(r.a1.z, r.a1.w);
    *(int4*)&As_p[buf][a_r][a_kp0] = p;
    Bs_p[buf][b_c0 + 0][b_kp] = pack_bf16x2(r.u0.x, r.w0.x);
    Bs_p[buf][b_c0 + 1][b_kp] = pack_bf16x2(r.u0.y, r.w0.y);
    Bs_p[buf][b_c0 + 2][b_kp] = pack_bf16x2(r.u0.z, r.w0.z);
    Bs_p[buf][b_c0 + 3][b_kp] = pack_bf16x2(r.u0.w, r.w0.w);
    Bs_p[buf][b_c0 + 4][b_kp] = pack_bf16x2(r.u1.x, r.w1.x);
    Bs_p[buf][b_c0 + 5][b_kp] = pack_bf16x2(r.u1.y, r.w1.y);
    Bs_p[buf][b_c0 + 6][b_kp] = pack_bf16x2(r.u1.z, r.w1.z);
    Bs_p[buf][b_c0 + 7][b_kp] = pack_bf16x2(r.u1.w, r.w1.w);
  };

  // prologue: stage tile 0 into buffer 0
  {
    TileRegs r;
    load_tile(kBeg, r);
    stage_tile(0, r);
  }
  __syncthreads();

  for (int it = 0; it < nIter; ++it) {
    const int cur = it & 1;

    // prefetch next tile into registers (overlaps with the WMMA work below)
    TileRegs nxt;
    if (it + 1 < nIter) load_tile(kBeg + (it + 1) * 32, nxt);

    // fragments: A -> kp [4kh..+3],[8+4kh..+3]; B(col) -> kp [4kh..+3],[8+4kh..+3]
    FragU fa, fb0, fb1, fb2, fb3;
    {
      const unsigned int* ar = As_p[cur][wm * 16 + m_l];
      fa.i[0] = *(const int4*)(ar + 4 * kh);
      fa.i[1] = *(const int4*)(ar + 8 + 4 * kh);
      const int colb = wn * 64 + m_l;
      const unsigned int* b0 = Bs_p[cur][colb +  0];
      const unsigned int* b1 = Bs_p[cur][colb + 16];
      const unsigned int* b2 = Bs_p[cur][colb + 32];
      const unsigned int* b3 = Bs_p[cur][colb + 48];
      fb0.i[0] = *(const int4*)(b0 + 4 * kh); fb0.i[1] = *(const int4*)(b0 + 8 + 4 * kh);
      fb1.i[0] = *(const int4*)(b1 + 4 * kh); fb1.i[1] = *(const int4*)(b1 + 8 + 4 * kh);
      fb2.i[0] = *(const int4*)(b2 + 4 * kh); fb2.i[1] = *(const int4*)(b2 + 8 + 4 * kh);
      fb3.i[0] = *(const int4*)(b3 + 4 * kh); fb3.i[1] = *(const int4*)(b3 + 8 + 4 * kh);
    }
    acc0 = __builtin_amdgcn_wmma_f32_16x16x32_bf16(false, fa.v, false, fb0.v, (short)0, acc0, false, false);
    acc1 = __builtin_amdgcn_wmma_f32_16x16x32_bf16(false, fa.v, false, fb1.v, (short)0, acc1, false, false);
    acc2 = __builtin_amdgcn_wmma_f32_16x16x32_bf16(false, fa.v, false, fb2.v, (short)0, acc2, false, false);
    acc3 = __builtin_amdgcn_wmma_f32_16x16x32_bf16(false, fa.v, false, fb3.v, (short)0, acc3, false, false);

    // stage prefetched tile into the other buffer (safe: prior barrier ensured
    // everyone finished reading it in iteration it-1)
    if (it + 1 < nIter) stage_tile(cur ^ 1, nxt);
    __syncthreads();
  }

  // epilogue: C/D layout -> lane n = lane%16, row m = v + 8*(lane/16)
  if (gridDim.z == 1) {
    #pragma unroll
    for (int i = 0; i < 4; ++i) {
      v8f accv = (i == 0) ? acc0 : (i == 1) ? acc1 : (i == 2) ? acc2 : acc3;
      int col = colBase + wn * 64 + i * 16 + m_l;
      float bv = bias[col];
      #pragma unroll
      for (int v = 0; v < 8; ++v) {
        int row = rowBase + wm * 16 + v + 8 * kh;
        float x = accv[v] + bv;
        C[(long long)row * N + col] = x > 0.f ? x : 0.f;
      }
    }
  } else {
    #pragma unroll
    for (int i = 0; i < 4; ++i) {
      v8f accv = (i == 0) ? acc0 : (i == 1) ? acc1 : (i == 2) ? acc2 : acc3;
      int col = colBase + wn * 64 + i * 16 + m_l;
      #pragma unroll
      for (int v = 0; v < 8; ++v) {
        int row = rowBase + wm * 16 + v + 8 * kh;
        atomicAdd(&C[(long long)row * N + col], accv[v]);
      }
    }
  }
}

// split-K epilogue: C = relu(C + bias)
__global__ __launch_bounds__(256) void bias_relu_kernel(
    float* __restrict__ C, const float* __restrict__ bias, int N, int total)
{
  int i = blockIdx.x * 256 + threadIdx.x;
  if (i < total) {
    float x = C[i] + bias[i & (N - 1)];   // N is a power of two here
    C[i] = x > 0.f ? x : 0.f;
  }
}

// ---------------------------------------------------------------------------
// Fused VQ: per 128 rows -> recompute token embedding e into LDS, f32-WMMA
// distance GEMM vs all 512 codes, running argmin, gather q=E[idx], counts
// histogram, (q-e)^2 loss accumulation. No dist matrix, no stored e.
// ---------------------------------------------------------------------------
__global__ __launch_bounds__(256) void vq_kernel(
    const float* __restrict__ zenc,   // [262144]
    const float* __restrict__ E,      // [512][64]
    const float* __restrict__ Wt1, const float* __restrict__ bt1,   // [32],[32]
    const float* __restrict__ Wt2, const float* __restrict__ bt2,   // [32][64],[64]
    float* __restrict__ Q,            // [262144][64]  (q_st output slot)
    float* __restrict__ counts,       // [512]
    float* __restrict__ loss_acc)     // [1]
{
  __shared__ float es[128][66];       // e rows (+2 pad)
  __shared__ float w1s[32], b1s[32], b2s[64];
  __shared__ float w2s[32][64];
  __shared__ float esq[512];
  __shared__ float pv[8][16][16];
  __shared__ int   pi[8][16][16];
  __shared__ float red[128];

  const int t = threadIdx.x;
  const long long rowBlock = (long long)blockIdx.x * 128;

  if (t < 32) { w1s[t] = Wt1[t]; b1s[t] = bt1[t]; }
  if (t < 64) b2s[t] = bt2[t];
  for (int i = t; i < 32 * 64; i += 256) w2s[i >> 6][i & 63] = Wt2[i];
  for (int c = t; c < 512; c += 256) {
    float s = 0.f;
    const float* er = E + c * 64;
    #pragma unroll 8
    for (int j = 0; j < 64; ++j) { float v = er[j]; s += v * v; }
    esq[c] = s;
  }
  __syncthreads();

  // token embedding: 2 threads per row, 32 outputs each
  {
    int r = t >> 1;
    int half = (t & 1) * 32;
    float z = zenc[rowBlock + r];
    float acc[32];
    #pragma unroll
    for (int j = 0; j < 32; ++j) acc[j] = b2s[half + j];
    #pragma unroll 4
    for (int i = 0; i < 32; ++i) {
      float h = z * w1s[i] + b1s[i];
      h = h > 0.f ? h : 0.f;
      #pragma unroll
      for (int j = 0; j < 32; ++j) acc[j] += h * w2s[i][half + j];
    }
    #pragma unroll
    for (int j = 0; j < 32; ++j) es[r][half + j] = acc[j];
  }
  __syncthreads();

  const int lane = t & 31;
  const int wave = t >> 5;            // wave handles rows wave*16..+15
  const int m_l  = lane & 15;
  const int kh   = lane >> 4;

  // preload all 16 f32 A fragments (16x4 layout: VGPR0 K=0/2, VGPR1 K=1/3)
  v2f af[16];
  #pragma unroll
  for (int s = 0; s < 16; ++s) {
    af[s].x = es[wave * 16 + m_l][4 * s + 2 * kh];
    af[s].y = es[wave * 16 + m_l][4 * s + 2 * kh + 1];
  }

  float bv[8]; int bi[8];
  #pragma unroll
  for (int v = 0; v < 8; ++v) { bv[v] = 3.4e38f; bi[v] = 0; }

  for (int nt = 0; nt < 32; ++nt) {
    v8f acc = {};
    const float* Ebase = E + (nt * 16 + m_l) * 64 + 2 * kh;
    #pragma unroll
    for (int s = 0; s < 16; ++s) {
      v2f b;
      b.x = Ebase[4 * s];
      b.y = Ebase[4 * s + 1];
      acc = __builtin_amdgcn_wmma_f32_16x16x4_f32(false, af[s], false, b, (short)0, acc, false, false);
    }
    int code = nt * 16 + m_l;
    float eq = esq[code];
    #pragma unroll
    for (int v = 0; v < 8; ++v) {
      float dist = eq - 2.f * acc[v];           // argmin(||E||^2 - 2 e.E)
      if (dist < bv[v]) { bv[v] = dist; bi[v] = code; }
    }
  }
  #pragma unroll
  for (int v = 0; v < 8; ++v) {
    int m = v + 8 * kh;
    pv[wave][m][m_l] = bv[v];
    pi[wave][m][m_l] = bi[v];
  }
  __syncthreads();

  float lsum = 0.f;
  if (t < 128) {
    int w = t >> 4, m = t & 15;
    float best = pv[w][m][0]; int bidx = pi[w][m][0];
    #pragma unroll
    for (int n = 1; n < 16; ++n) {
      float d = pv[w][m][n]; int c = pi[w][m][n];
      if (d < best || (d == best && c < bidx)) { best = d; bidx = c; }
    }
    long long row = rowBlock + w * 16 + m;
    atomicAdd(&counts[bidx], 1.0f);
    const float* er = E + bidx * 64;
    float* qr = Q + row * 64;
    #pragma unroll 8
    for (int j = 0; j < 64; ++j) {
      float ev = er[j];
      qr[j] = ev;
      float dd = ev - es[w * 16 + m][j];
      lsum += dd * dd;
    }
    red[t] = lsum;
  }
  __syncthreads();
  for (int s = 64; s >= 1; s >>= 1) {
    if (t < s) red[t] += red[t + s];
    __syncthreads();
  }
  if (t == 0) atomicAdd(loss_acc, red[0]);
}

// ---------------------------------------------------------------------------
// Residual block: y = x + BN(relu(relu(x@W1+b1)@W2+b2))  (eval-mode BN)
// ---------------------------------------------------------------------------
__global__ __launch_bounds__(256) void resblock_kernel(
    const float* __restrict__ X, const float* __restrict__ W1,
    const float* __restrict__ b1, const float* __restrict__ W2,
    const float* __restrict__ b2, const float* __restrict__ gamma,
    const float* __restrict__ beta, float* __restrict__ Y)
{
  __shared__ float xs[512];
  __shared__ float ps[16][17];
  __shared__ float hs[16];
  const int t = threadIdx.x;
  const long long b = blockIdx.x;
  xs[t]       = X[b * 512 + t];
  xs[t + 256] = X[b * 512 + 256 + t];
  __syncthreads();
  {
    int j = t & 15, ch = t >> 4;
    int d0 = ch * 32;
    float s = 0.f;
    #pragma unroll 8
    for (int d = 0; d < 32; ++d) s += xs[d0 + d] * W1[(d0 + d) * 16 + j];
    ps[j][ch] = s;
  }
  __syncthreads();
  if (t < 16) {
    float s = b1[t];
    #pragma unroll
    for (int c = 0; c < 16; ++c) s += ps[t][c];
    hs[t] = s > 0.f ? s : 0.f;
  }
  __syncthreads();
  const float invs = 0.99999500003749978f;   // 1/sqrt(1 + 1e-5)
  for (int c = t; c < 512; c += 256) {
    float s = b2[c];
    #pragma unroll
    for (int j = 0; j < 16; ++j) s += hs[j] * W2[j * 512 + c];
    s = s > 0.f ? s : 0.f;
    s = s * invs * gamma[c] + beta[c];
    Y[b * 512 + c] = xs[c] + s;
  }
}

// ---------------------------------------------------------------------------
// Classifier head: sigmoid(q[b,:]@Wc + bc), N=10. One block per batch row.
// ---------------------------------------------------------------------------
__global__ __launch_bounds__(256) void cls_kernel(
    const float* __restrict__ Q, const float* __restrict__ Wc,
    const float* __restrict__ bc, float* __restrict__ out)
{
  __shared__ float red[256];
  const int t = threadIdx.x;
  const long long b = blockIdx.x;
  float acc[10];
  #pragma unroll
  for (int j = 0; j < 10; ++j) acc[j] = 0.f;
  const float* q = Q + b * 32768;
  for (int k = t; k < 32768; k += 256) {
    float qv = q[k];
    const float* wr = Wc + (long long)k * 10;
    #pragma unroll
    for (int j = 0; j < 10; ++j) acc[j] += qv * wr[j];
  }
  for (int j = 0; j < 10; ++j) {
    red[t] = acc[j];
    __syncthreads();
    for (int s = 128; s >= 1; s >>= 1) {
      if (t < s) red[t] += red[t + s];
      __syncthreads();
    }
    if (t == 0) {
      float x = red[0] + bc[j];
      out[b * 10 + j] = 1.f / (1.f + __expf(-x));
    }
    __syncthreads();
  }
}

// ---------------------------------------------------------------------------
// Finalize: perplexity from counts, vq_loss = 1.25 * sum / (262144*64)
// ---------------------------------------------------------------------------
__global__ void finalize_kernel(const float* __restrict__ counts,
                                const float* __restrict__ loss_acc,
                                float* __restrict__ out_loss,
                                float* __restrict__ out_perp)
{
  __shared__ float red[512];
  const int t = threadIdx.x;   // 512 threads
  float p = counts[t] * (1.0f / 262144.0f);
  red[t] = -p * __logf(p + 1e-10f);
  __syncthreads();
  for (int s = 256; s >= 1; s >>= 1) {
    if (t < s) red[t] += red[t + s];
    __syncthreads();
  }
  if (t == 0) {
    *out_perp = __expf(red[0]);
    *out_loss = loss_acc[0] * 1.25f / (262144.0f * 64.0f);
  }
}

// ---------------------------------------------------------------------------
extern "C" void kernel_launch(void* const* d_in, const int* in_sizes, int n_in,
                              void* d_out, int out_size, void* d_ws, size_t ws_size,
                              hipStream_t stream)
{
  const float* x    = (const float*)d_in[0];
  const float* We1  = (const float*)d_in[1];
  const float* be1  = (const float*)d_in[2];
  const float* We2  = (const float*)d_in[3];
  const float* be2  = (const float*)d_in[4];
  const float* Wre1 = (const float*)d_in[5];
  const float* bre1 = (const float*)d_in[6];
  const float* Wre2 = (const float*)d_in[7];
  const float* bre2 = (const float*)d_in[8];
  const float* ge   = (const float*)d_in[9];
  const float* bebn = (const float*)d_in[10];
  const float* E    = (const float*)d_in[11];
  const float* Wt1  = (const float*)d_in[12];
  const float* bt1  = (const float*)d_in[13];
  const float* Wt2  = (const float*)d_in[14];
  const float* bt2  = (const float*)d_in[15];
  const float* Wc   = (const float*)d_in[16];
  const float* bc   = (const float*)d_in[17];
  const float* Wp   = (const float*)d_in[18];
  const float* bp   = (const float*)d_in[19];
  const float* Wrd1 = (const float*)d_in[20];
  const float* brd1 = (const float*)d_in[21];
  const float* Wrd2 = (const float*)d_in[22];
  const float* brd2 = (const float*)d_in[23];
  const float* gd   = (const float*)d_in[24];
  const float* bdbn = (const float*)d_in[25];
  const float* Wd1  = (const float*)d_in[26];
  const float* bd1  = (const float*)d_in[27];
  const float* Wd2  = (const float*)d_in[28];
  const float* bd2  = (const float*)d_in[29];

  // d_out layout: [vq_loss(1)][x_recon(512*4096)][perp(1)][q_st(512*32768)][cls(5120)][zenc(262144)]
  float* out    = (float*)d_out;
  float* o_xrec = out + 1;
  float* o_perp = out + 2097153;
  float* o_qst  = out + 2097154;
  float* o_cls  = out + 18874370;
  float* o_zenc = out + 18879490;

  // workspace layout (floats)
  float* ws     = (float*)d_ws;
  float* h      = ws;                  // 1,048,576  (also reused as d2)
  float* zpre   = ws + 1048576;        // 262,144
  float* d1     = zpre + 262144;       // 262,144
  float* dres   = d1 + 262144;         // 262,144
  float* counts = dres + 262144;       // 512
  float* lossa  = counts + 512;        // 1

  hipMemsetAsync(counts, 0, (512 + 1) * sizeof(float), stream);
  hipMemsetAsync(zpre, 0, 262144 * sizeof(float), stream);   // split-K accumulators
  hipMemsetAsync(d1,   0, 262144 * sizeof(float), stream);

  dim3 blk(256);
  // encoder
  gemm_bias_relu<<<dim3(16, 8, 1), blk, 0, stream>>>(x, We1, be1, h, 512, 2048, 4096);
  gemm_bias_relu<<<dim3(4, 8, 4), blk, 0, stream>>>(h, We2, be2, zpre, 512, 512, 2048);
  bias_relu_kernel<<<1024, blk, 0, stream>>>(zpre, be2, 512, 262144);
  resblock_kernel<<<512, blk, 0, stream>>>(zpre, Wre1, bre1, Wre2, bre2, ge, bebn, o_zenc);
  // vector quantizer (embed + dist + argmin + gather + counts + loss)
  vq_kernel<<<2048, blk, 0, stream>>>(o_zenc, E, Wt1, bt1, Wt2, bt2, o_qst, counts, lossa);
  // decoder (q_st forward value == q); split-K=8 for the K=32768 GEMM
  gemm_bias_relu<<<dim3(4, 8, 8), blk, 0, stream>>>(o_qst, Wp, bp, d1, 512, 512, 32768);
  bias_relu_kernel<<<1024, blk, 0, stream>>>(d1, bp, 512, 262144);
  resblock_kernel<<<512, blk, 0, stream>>>(d1, Wrd1, brd1, Wrd2, brd2, gd, bdbn, dres);
  gemm_bias_relu<<<dim3(16, 8, 1), blk, 0, stream>>>(dres, Wd1, bd1, h, 512, 2048, 512);
  gemm_bias_relu<<<dim3(32, 8, 1), blk, 0, stream>>>(h, Wd2, bd2, o_xrec, 512, 4096, 2048);
  // classifier + scalars
  cls_kernel<<<512, blk, 0, stream>>>(o_qst, Wc, bc, o_cls);
  finalize_kernel<<<1, 512, 0, stream>>>(counts, lossa, out, o_perp);
}
